// GraphormerLayerV2_71159018160677
// MI455X (gfx1250) — matrix operand order repaired and checked by reading, not compiled, not run on hardware
//
#include <hip/hip_runtime.h>

typedef __bf16 bf16;
typedef __attribute__((ext_vector_type(16))) __bf16 v16bf;
typedef __attribute__((ext_vector_type(8)))  __bf16 v8bf;
typedef __attribute__((ext_vector_type(8)))  float  v8f;

// exact parameter vector type of the async-LDS builtins (per hipcc diagnostic)
typedef int gv4i __attribute__((vector_size(4 * sizeof(int))));
typedef __attribute__((address_space(1))) gv4i* gv4i_g;
typedef __attribute__((address_space(3))) gv4i* gv4i_l;

union BFrag16 { v16bf v; v8bf h[2]; };

static __device__ __forceinline__ bf16 f2bf(float x) { return (bf16)x; }

#define NB   8
#define NSEQ 512
#define DIM  1024
#define NH   16
#define HD   64
#define MROWS (NB*NSEQ)   // 4096

// ---------------------------------------------------------------------------
// CDNA5 async global->LDS copy (ASYNCcnt path) with sync fallback.
// ---------------------------------------------------------------------------
static __device__ __forceinline__ void async_ld16(void* lds, const void* g)
{
#if __has_builtin(__builtin_amdgcn_global_load_async_to_lds_b128)
    __builtin_amdgcn_global_load_async_to_lds_b128(
        (gv4i_g)(gv4i*)g, (gv4i_l)(gv4i*)lds, 0, 0);
#else
    *(v8bf*)lds = *(const v8bf*)g;
#endif
}

static __device__ __forceinline__ void wait_async()
{
#if __has_builtin(__builtin_amdgcn_s_wait_asynccnt)
    __builtin_amdgcn_s_wait_asynccnt(0);
#else
    asm volatile("s_wait_asynccnt 0" ::: "memory");
#endif
}

// ---------------------------------------------------------------------------
// Weight transpose + cast: out[n][k] = bf16(in[k][n]); in is K x N row-major.
// ---------------------------------------------------------------------------
__global__ void __launch_bounds__(256)
k_transpose_cast(const float* __restrict__ in, bf16* __restrict__ out, int K, int N)
{
    __shared__ float tile[32][33];
    const int n0 = blockIdx.x * 32, k0 = blockIdx.y * 32;
    const int tx = threadIdx.x, ty = threadIdx.y;        // 32 x 8
    #pragma unroll
    for (int j = 0; j < 32; j += 8)
        tile[ty + j][tx] = in[(size_t)(k0 + ty + j) * N + n0 + tx];
    __syncthreads();
    #pragma unroll
    for (int j = 0; j < 32; j += 8)
        out[(size_t)(n0 + ty + j) * K + k0 + tx] = f2bf(tile[tx][ty + j]);
}

__global__ void __launch_bounds__(256)
k_cast(const float* __restrict__ in, bf16* __restrict__ out, int n)
{
    int i = blockIdx.x * 256 + threadIdx.x;
    if (i < n) out[i] = f2bf(in[i]);
}

// ---------------------------------------------------------------------------
// bf16 WMMA GEMM:  C[M][Ncols] = A[M][K] * Bt[Ncols][K]^T + bias
// Block tile 128x128, 8 waves each computing 32(M) x 64(N).
// K-step 64, double-buffered LDS filled by async global->LDS copies.
// ---------------------------------------------------------------------------
enum { EP_F32 = 0, EP_BF16 = 1, EP_GELU_BF16 = 2 };

template <int MODE>
__global__ void __launch_bounds__(256)
k_gemm(const bf16* __restrict__ A, const bf16* __restrict__ Bt,
       const float* __restrict__ bias, float* __restrict__ outF,
       bf16* __restrict__ outB, int M, int Ncols, int K)
{
    constexpr int KSTEP = 64;
    constexpr int LDT   = 72;               // padded LDS row stride (bf16 elems)
    constexpr int BUFE  = 128 * LDT;        // elems per buffer
    __shared__ __align__(16) bf16 lA[2 * BUFE];
    __shared__ __align__(16) bf16 lB[2 * BUFE];

    const int tid  = threadIdx.x;
    const int wave = tid >> 5, lane = tid & 31;
    const int hl   = lane >> 4, l16 = lane & 15;
    const int wm   = wave & 3,  wn  = wave >> 2;
    const int m0   = blockIdx.y * 128, n0 = blockIdx.x * 128;

    v8f acc[2][4];
    #pragma unroll
    for (int mt = 0; mt < 2; mt++)
        #pragma unroll
        for (int nt = 0; nt < 4; nt++) acc[mt][nt] = (v8f){};

    // issue async copies of one 128xKSTEP tile pair into buffer p
    auto issue = [&](int p, int k0) {
        bf16* pA = lA + p * BUFE;
        bf16* pB = lB + p * BUFE;
        #pragma unroll
        for (int i = 0; i < 4; i++) {
            int idx = tid + i * 256;          // 0..1023: 128 rows x 8 quads of 8 bf16
            int row = idx >> 3, q = idx & 7;
            async_ld16(&pA[row * LDT + q * 8], &A [(size_t)(m0 + row) * K + k0 + q * 8]);
            async_ld16(&pB[row * LDT + q * 8], &Bt[(size_t)(n0 + row) * K + k0 + q * 8]);
        }
    };

    issue(0, 0);
    int p = 0;
    for (int k0 = 0; k0 < K; k0 += KSTEP) {
        wait_async();                 // own copies into buffer p landed
        __syncthreads();              // ... for every wave; and p^1 free to refill
        if (k0 + KSTEP < K) issue(p ^ 1, k0 + KSTEP);

        const bf16* bufA = lA + p * BUFE;
        const bf16* bufB = lB + p * BUFE;
        #pragma unroll
        for (int kc = 0; kc < 2; kc++) {
            const int kof = kc * 32;
            BFrag16 af[2], bfr[4];
            #pragma unroll
            for (int mt = 0; mt < 2; mt++) {
                int r = wm * 32 + mt * 16 + l16;
                af[mt].h[0] = *(const v8bf*)&bufA[r * LDT + kof + hl * 8];       // K=hl*8..+7
                af[mt].h[1] = *(const v8bf*)&bufA[r * LDT + kof + hl * 8 + 16];  // K=16+hl*8..
            }
            #pragma unroll
            for (int nt = 0; nt < 4; nt++) {
                int r = wn * 64 + nt * 16 + l16;
                bfr[nt].h[0] = *(const v8bf*)&bufB[r * LDT + kof + hl * 8];      // K=hl*8..+7
                bfr[nt].h[1] = *(const v8bf*)&bufB[r * LDT + kof + hl * 8 + 16];
            }
            #pragma unroll
            for (int mt = 0; mt < 2; mt++)
                #pragma unroll
                for (int nt = 0; nt < 4; nt++)
                    acc[mt][nt] = __builtin_amdgcn_wmma_f32_16x16x32_bf16(
                        false, af[mt].v, false, bfr[nt].v,
                        (short)0, acc[mt][nt], false, false);
        }
        __syncthreads();
        p ^= 1;
    }

    #pragma unroll
    for (int mt = 0; mt < 2; mt++)
        #pragma unroll
        for (int nt = 0; nt < 4; nt++) {
            int col = n0 + wn * 64 + nt * 16 + l16;
            float bv = bias[col];
            #pragma unroll
            for (int i = 0; i < 8; i++) {
                int row = m0 + wm * 32 + mt * 16 + i + hl * 8;
                float v = acc[mt][nt][i] + bv;
                if (MODE == EP_F32) {
                    outF[(size_t)row * Ncols + col] = v;
                } else if (MODE == EP_BF16) {
                    outB[(size_t)row * Ncols + col] = f2bf(v);
                } else {
                    float g = 0.5f * v * (1.0f + erff(v * 0.70710678118654752f));
                    outB[(size_t)row * Ncols + col] = f2bf(g);
                }
            }
        }
}

// ---------------------------------------------------------------------------
// qkv[M][3072] bf16  ->  q[b][h][n][d], k[b][h][n][d], vT[b][h][d][n]  (bf16)
// ---------------------------------------------------------------------------
__global__ void __launch_bounds__(256)
k_reshape(const bf16* __restrict__ qkv, bf16* __restrict__ qh,
          bf16* __restrict__ kh, bf16* __restrict__ vt)
{
    int i   = blockIdx.x * 256 + threadIdx.x;   // 0 .. 4M-1
    int m   = i >> 10, col = i & 1023;
    int b   = m >> 9,  n   = m & 511;
    int h   = col >> 6, d  = col & 63;
    int bh  = b * NH + h;
    size_t base = (size_t)m * 3072;
    qh[((size_t)bh * NSEQ + n) * HD + d] = qkv[base + col];
    kh[((size_t)bh * NSEQ + n) * HD + d] = qkv[base + 1024 + col];
    vt[((size_t)bh * HD + d) * NSEQ + n] = qkv[base + 2048 + col];
}

// ---------------------------------------------------------------------------
// Flash-style attention. 8 waves/block; each wave owns one 16-query tile.
// ---------------------------------------------------------------------------
__global__ void __launch_bounds__(256)
k_attn(const bf16* __restrict__ qh, const bf16* __restrict__ kh,
       const bf16* __restrict__ vt, const float* __restrict__ bias,
       const unsigned char* __restrict__ mask, bf16* __restrict__ outB)
{
    __shared__ __align__(16) bf16 ldsP[8 * 512];      // 16x32 bf16 P-tile per wave

    const int tid  = threadIdx.x;
    const int wave = tid >> 5, lane = tid & 31;
    const int hl   = lane >> 4, l16 = lane & 15;

    const int qt = blockIdx.x * 8 + wave;      // global 16-row query tile id
    const int bh = qt >> 5;                    // 32 tiles per (b,h)
    const int q0 = (qt & 31) * 16;
    const int b  = bh >> 4, h = bh & 15;

    const bf16* qp = qh + ((size_t)bh * NSEQ + q0) * HD;
    const bf16* kp = kh + (size_t)bh * NSEQ * HD;
    const bf16* vp = vt + (size_t)bh * HD * NSEQ;
    const float* bp = bias + ((size_t)bh * NSEQ + q0) * NSEQ;
    const unsigned char* mp = mask + b * NSEQ;
    bf16* lp = ldsP + wave * 512;

    // Q as A-fragments: two K-chunks of 32 features each
    BFrag16 aq[2];
    #pragma unroll
    for (int c = 0; c < 2; c++) {
        aq[c].h[0] = *(const v8bf*)&qp[l16 * HD + c * 32 + hl * 8];
        aq[c].h[1] = *(const v8bf*)&qp[l16 * HD + c * 32 + hl * 8 + 16];
    }

    v8f accO[4];
    #pragma unroll
    for (int dt = 0; dt < 4; dt++) accO[dt] = (v8f){};
    float mrun[8], lrun[8];
    #pragma unroll
    for (int i = 0; i < 8; i++) { mrun[i] = -__builtin_inff(); lrun[i] = 0.0f; }

    for (int kb = 0; kb < NSEQ; kb += 32) {
        v8f s[2];
        #pragma unroll
        for (int t = 0; t < 2; t++) {
            const int key16 = kb + t * 16;
            const bf16* krow = kp + (size_t)(key16 + l16) * HD;
            BFrag16 bk0, bk1;
            bk0.h[0] = *(const v8bf*)&krow[hl * 8];
            bk0.h[1] = *(const v8bf*)&krow[hl * 8 + 16];
            bk1.h[0] = *(const v8bf*)&krow[32 + hl * 8];
            bk1.h[1] = *(const v8bf*)&krow[32 + hl * 8 + 16];
            v8f z = (v8f){};
            z = __builtin_amdgcn_wmma_f32_16x16x32_bf16(false, aq[0].v, false, bk0.v,
                                                        (short)0, z, false, false);
            z = __builtin_amdgcn_wmma_f32_16x16x32_bf16(false, aq[1].v, false, bk1.v,
                                                        (short)0, z, false, false);
            const unsigned char mv = mp[key16 + l16];
            #pragma unroll
            for (int i = 0; i < 8; i++) {
                float vb  = bp[(size_t)(i + hl * 8) * NSEQ + key16 + l16];
                float val = z[i] * 0.125f + vb;           // 1/sqrt(64)
                z[i] = mv ? val : -__builtin_inff();
            }
            s[t] = z;
        }

        // online softmax over 32 new keys
        float corr[8];
        #pragma unroll
        for (int i = 0; i < 8; i++) {
            float t = fmaxf(s[0][i], s[1][i]);
            #pragma unroll
            for (int off = 1; off < 16; off <<= 1)
                t = fmaxf(t, __shfl_xor(t, off, 32));     // halves stay separate
            float mn = fmaxf(mrun[i], t);
            float p0 = expf(s[0][i] - mn);
            float p1 = expf(s[1][i] - mn);
            float rs = p0 + p1;
            #pragma unroll
            for (int off = 1; off < 16; off <<= 1)
                rs += __shfl_xor(rs, off, 32);
            float c = expf(mrun[i] - mn);
            lrun[i] = lrun[i] * c + rs;
            mrun[i] = mn;
            corr[i] = c;
            lp[(size_t)(i + hl * 8) * 32 + l16]      = f2bf(p0);
            lp[(size_t)(i + hl * 8) * 32 + 16 + l16] = f2bf(p1);
        }
        asm volatile("s_wait_dscnt 0" ::: "memory");      // P visible wave-wide

        BFrag16 ap;
        ap.h[0] = *(const v8bf*)&lp[l16 * 32 + hl * 8];
        ap.h[1] = *(const v8bf*)&lp[l16 * 32 + hl * 8 + 16];

        #pragma unroll
        for (int dt = 0; dt < 4; dt++) {
            const bf16* vrow = vp + (size_t)(dt * 16 + l16) * NSEQ + kb + hl * 8;
            BFrag16 bv;
            bv.h[0] = *(const v8bf*)&vrow[0];
            bv.h[1] = *(const v8bf*)&vrow[16];
            #pragma unroll
            for (int i = 0; i < 8; i++) accO[dt][i] *= corr[i];
            accO[dt] = __builtin_amdgcn_wmma_f32_16x16x32_bf16(
                false, ap.v, false, bv.v, (short)0, accO[dt], false, false);
        }
    }

    #pragma unroll
    for (int dt = 0; dt < 4; dt++)
        #pragma unroll
        for (int i = 0; i < 8; i++) {
            int row = b * NSEQ + q0 + i + hl * 8;
            int col = h * HD + dt * 16 + l16;
            outB[(size_t)row * DIM + col] = f2bf(accO[dt][i] / lrun[i]);
        }
}

// ---------------------------------------------------------------------------
// y = LN(x + res_scale*delta) * g + b ; writes f32 and optional bf16 copy.
// One block per row of D=1024; 256 threads x 4 elems.
// ---------------------------------------------------------------------------
__global__ void __launch_bounds__(256)
k_ln(const float* __restrict__ xin, const float* __restrict__ delta,
     const float* __restrict__ rs, const float* __restrict__ g,
     const float* __restrict__ bta, float* __restrict__ outF,
     bf16* __restrict__ outB)
{
    __shared__ float ssum[8], ssq[8];
    const int row = blockIdx.x, tid = threadIdx.x;
    const float r = rs[0];
    const float* xr = xin   + (size_t)row * DIM;
    const float* dr = delta + (size_t)row * DIM;

    float v[4], s = 0.0f, sq = 0.0f;
    #pragma unroll
    for (int j = 0; j < 4; j++) {
        int c = tid + j * 256;
        float t = xr[c] + r * dr[c];
        v[j] = t; s += t; sq += t * t;
    }
    #pragma unroll
    for (int off = 16; off > 0; off >>= 1) {
        s  += __shfl_xor(s,  off, 32);
        sq += __shfl_xor(sq, off, 32);
    }
    if ((tid & 31) == 0) { ssum[tid >> 5] = s; ssq[tid >> 5] = sq; }
    __syncthreads();
    if (tid == 0) {
        float a = 0.0f, c2 = 0.0f;
        for (int j = 0; j < 8; j++) { a += ssum[j]; c2 += ssq[j]; }
        ssum[0] = a; ssq[0] = c2;
    }
    __syncthreads();
    const float mu  = ssum[0] * (1.0f / DIM);
    const float var = ssq[0] * (1.0f / DIM) - mu * mu;
    const float inv = rsqrtf(var + 1e-5f);
    #pragma unroll
    for (int j = 0; j < 4; j++) {
        int c = tid + j * 256;
        float y = (v[j] - mu) * inv * g[c] + bta[c];
        outF[(size_t)row * DIM + c] = y;
        if (outB) outB[(size_t)row * DIM + c] = f2bf(y);
    }
}

// ---------------------------------------------------------------------------
extern "C" void kernel_launch(void* const* d_in, const int* in_sizes, int n_in,
                              void* d_out, int out_size, void* d_ws, size_t ws_size,
                              hipStream_t stream)
{
    (void)in_sizes; (void)n_in; (void)out_size; (void)ws_size;
    const float* x     = (const float*)d_in[0];
    const float* abias = (const float*)d_in[1];
    const unsigned char* nmask = (const unsigned char*)d_in[2];
    const float* Wqkv  = (const float*)d_in[3];
    const float* bqkv  = (const float*)d_in[4];
    const float* Wout  = (const float*)d_in[5];
    const float* bout  = (const float*)d_in[6];
    const float* g1    = (const float*)d_in[7];
    const float* b1    = (const float*)d_in[8];
    const float* g2    = (const float*)d_in[9];
    const float* b2    = (const float*)d_in[10];
    const float* W1    = (const float*)d_in[11];
    const float* bf1   = (const float*)d_in[12];
    const float* W2    = (const float*)d_in[13];
    const float* bf2   = (const float*)d_in[14];
    const float* rsc   = (const float*)d_in[15];

    char* ws = (char*)d_ws;
    size_t off = 0;
    auto alloc = [&](size_t bytes) -> char* {
        char* p = ws + off;
        off = (off + bytes + 255) & ~(size_t)255;
        return p;
    };
    bf16*  wqkvT = (bf16*) alloc((size_t)3072 * 1024 * 2);
    bf16*  woutT = (bf16*) alloc((size_t)1024 * 1024 * 2);
    bf16*  w1T   = (bf16*) alloc((size_t)4096 * 1024 * 2);
    bf16*  w2T   = (bf16*) alloc((size_t)1024 * 4096 * 2);
    bf16*  xbf   = (bf16*) alloc((size_t)MROWS * 1024 * 2);
    bf16*  qkvbf = (bf16*) alloc((size_t)MROWS * 3072 * 2);
    bf16*  qhp   = (bf16*) alloc((size_t)MROWS * 1024 * 2);
    bf16*  khp   = (bf16*) alloc((size_t)MROWS * 1024 * 2);
    bf16*  vtp   = (bf16*) alloc((size_t)MROWS * 1024 * 2);
    bf16*  aout  = (bf16*) alloc((size_t)MROWS * 1024 * 2);
    float* proj  = (float*)alloc((size_t)MROWS * 1024 * 4);
    float* x1f   = (float*)alloc((size_t)MROWS * 1024 * 4);
    bf16*  x1b   = (bf16*) alloc((size_t)MROWS * 1024 * 2);
    bf16*  h1    = (bf16*) alloc((size_t)MROWS * 4096 * 2);
    float* ffn   = (float*)alloc((size_t)MROWS * 1024 * 4);

    dim3 tb(32, 8);
    k_transpose_cast<<<dim3(3072/32, 1024/32), tb, 0, stream>>>(Wqkv, wqkvT, 1024, 3072);
    k_transpose_cast<<<dim3(1024/32, 1024/32), tb, 0, stream>>>(Wout, woutT, 1024, 1024);
    k_transpose_cast<<<dim3(4096/32, 1024/32), tb, 0, stream>>>(W1,   w1T,   1024, 4096);
    k_transpose_cast<<<dim3(1024/32, 4096/32), tb, 0, stream>>>(W2,   w2T,   4096, 1024);
    k_cast<<<(MROWS * 1024) / 256, 256, 0, stream>>>(x, xbf, MROWS * 1024);

    // QKV projection
    k_gemm<EP_BF16><<<dim3(3072/128, MROWS/128), 256, 0, stream>>>(
        xbf, wqkvT, bqkv, nullptr, qkvbf, MROWS, 3072, 1024);
    k_reshape<<<(MROWS * 1024) / 256, 256, 0, stream>>>(qkvbf, qhp, khp, vtp);

    // Attention (4096 q-tiles / 8 waves per block)
    k_attn<<<512, 256, 0, stream>>>(qhp, khp, vtp, abias, nmask, aout);

    // Output projection + LN1
    k_gemm<EP_F32><<<dim3(1024/128, MROWS/128), 256, 0, stream>>>(
        aout, woutT, bout, proj, nullptr, MROWS, 1024, 1024);
    k_ln<<<MROWS, 256, 0, stream>>>(x, proj, rsc, g1, b1, x1f, x1b);

    // FFN
    k_gemm<EP_GELU_BF16><<<dim3(4096/128, MROWS/128), 256, 0, stream>>>(
        x1b, w1T, bf1, nullptr, h1, MROWS, 4096, 1024);
    k_gemm<EP_F32><<<dim3(1024/128, MROWS/128), 256, 0, stream>>>(
        h1, w2T, bf2, ffn, nullptr, MROWS, 1024, 4096);

    // LN2 -> final output
    k_ln<<<MROWS, 256, 0, stream>>>(x1f, ffn, rsc, g2, b2, (float*)d_out, nullptr);
}
